// Model_27109833573061
// MI455X (gfx1250) — matrix operand order repaired and checked by reading, not compiled, not run on hardware
//
#include <hip/hip_runtime.h>
#include <hip/hip_bf16.h>

// ---- problem constants (match reference) ----
#define NNODES 10000
#define NEDGES 160000
#define ETOT   (NEDGES + NNODES)   // edges + self loops
#define DFEAT  1024
#define NHEAD  4
#define DHEAD  256
#define GGRAPH 64
#define NEG_SLOPE 0.2f

typedef __attribute__((ext_vector_type(16))) __bf16 v16bf;
typedef __attribute__((ext_vector_type(8)))  float  v8f;

// ---------------- conversion kernels ----------------
__global__ __launch_bounds__(256) void k_cvt_x(const float* __restrict__ x, __bf16* __restrict__ xb) {
  int i = blockIdx.x * 256 + threadIdx.x;
  if (i < NNODES * DFEAT) xb[i] = (__bf16)x[i];
}

// W2 is [K=1024, N=1024] row-major; produce Wt [N, K] (column-major of W2) in bf16
__global__ __launch_bounds__(256) void k_cvt_w(const float* __restrict__ W2, __bf16* __restrict__ wt) {
  int i = blockIdx.x * 256 + threadIdx.x;
  if (i < DFEAT * DFEAT) {
    int k = i >> 10, n = i & 1023;
    wt[(size_t)n * DFEAT + k] = (__bf16)W2[i];
  }
}

// ---------------- GEMM: h = x @ W2 via v_wmma_f32_16x16x32_bf16 ----------------
// block = 256 threads = 8 waves; wave tile = 16(M) x 64(N).
// Double-buffered fragments: the next K-half's 10 b128 loads are issued into
// the idle buffer before the current half's 4 WMMAs, so VMEM stays in flight
// across the matrix ops (partial s_wait_loadcnt instead of full drains).
__global__ __launch_bounds__(256) void k_gemm(const __bf16* __restrict__ A,
                                              const __bf16* __restrict__ Bt,
                                              float* __restrict__ Hout) {
  const int wave = threadIdx.x >> 5;
  const int lane = threadIdx.x & 31;
  const int mtile = blockIdx.x >> 1;
  const int colbase = (blockIdx.x & 1) * 512 + wave * 64;   // 4 x 16-wide tiles
  const int l15 = lane & 15;
  const int hi  = lane >> 4;                 // half-wave select

  // A fragment (16x32 bf16): lane holds row = l15; contiguous K run
  const __bf16* arow = A + (size_t)(mtile * 16 + l15) * DFEAT + hi * 16;
  // B fragments (32x16 bf16 each): lane holds col; contiguous K run (transposed W)
  const __bf16* brow0 = Bt + (size_t)(colbase +  0 + l15) * DFEAT + hi * 16;
  const __bf16* brow1 = Bt + (size_t)(colbase + 16 + l15) * DFEAT + hi * 16;
  const __bf16* brow2 = Bt + (size_t)(colbase + 32 + l15) * DFEAT + hi * 16;
  const __bf16* brow3 = Bt + (size_t)(colbase + 48 + l15) * DFEAT + hi * 16;

  typedef union { uint4 u[2]; v16bf v; } U;
  U aX, b0X, b1X, b2X, b3X;   // buffer X
  U aY, b0Y, b1Y, b2Y, b3Y;   // buffer Y

#define LDA(dst, k)      { (dst).u[0] = *(const uint4*)(arow + (k));  (dst).u[1] = *(const uint4*)(arow + (k) + 8); }
#define LDB(dst, p, k)   { (dst).u[0] = *(const uint4*)((p) + (k));   (dst).u[1] = *(const uint4*)((p) + (k) + 8); }
#define LDALL(s, k)      { LDA(a##s, (k)); LDB(b0##s, brow0, (k)); LDB(b1##s, brow1, (k)); LDB(b2##s, brow2, (k)); LDB(b3##s, brow3, (k)); }
#define MMALL(s)                                                                                           \
  { acc0 = __builtin_amdgcn_wmma_f32_16x16x32_bf16(false, a##s.v, false, b0##s.v, (short)0, acc0, false, false); \
    acc1 = __builtin_amdgcn_wmma_f32_16x16x32_bf16(false, a##s.v, false, b1##s.v, (short)0, acc1, false, false); \
    acc2 = __builtin_amdgcn_wmma_f32_16x16x32_bf16(false, a##s.v, false, b2##s.v, (short)0, acc2, false, false); \
    acc3 = __builtin_amdgcn_wmma_f32_16x16x32_bf16(false, a##s.v, false, b3##s.v, (short)0, acc3, false, false); }

  v8f acc0 = {}, acc1 = {}, acc2 = {}, acc3 = {};

  LDALL(X, 0);                       // prologue
#pragma unroll 1
  for (int k = 0; k < DFEAT; k += 64) {
    LDALL(Y, k + 32);                // prefetch next half into idle buffer
    MMALL(X);                        // compute on current buffer
    LDALL(X, k + 64);                // prefetch half after (last iter: pad read, unused)
    MMALL(Y);
  }
#undef LDA
#undef LDB
#undef LDALL
#undef MMALL

  // D layout: element v -> (m = v + 8*hi, n = l15)
  float* out = Hout + (size_t)(mtile * 16 + hi * 8) * DFEAT + colbase + l15;
#pragma unroll
  for (int v = 0; v < 8; ++v) {
    out[(size_t)v * DFEAT +  0] = acc0[v];
    out[(size_t)v * DFEAT + 16] = acc1[v];
    out[(size_t)v * DFEAT + 32] = acc2[v];
    out[(size_t)v * DFEAT + 48] = acc3[v];
  }
}

// ---------------- per-node attention logits es/ed ----------------
// one wave per (node, head): 4 waves / block
__global__ __launch_bounds__(128) void k_logits(const float* __restrict__ H,
                                                const float* __restrict__ asrc,
                                                const float* __restrict__ adst,
                                                float* __restrict__ es, float* __restrict__ ed) {
  int n = blockIdx.x;
  int head = threadIdx.x >> 5, lane = threadIdx.x & 31;
  const float* hrow = H + (size_t)n * DFEAT + head * DHEAD;
  const float* as = asrc + head * DHEAD;
  const float* ad = adst + head * DHEAD;
  float s0 = 0.f, s1 = 0.f;
#pragma unroll
  for (int j = lane; j < DHEAD; j += 32) {
    float hv = hrow[j];
    s0 += hv * as[j];
    s1 += hv * ad[j];
  }
  for (int off = 16; off; off >>= 1) {
    s0 += __shfl_xor(s0, off, 32);
    s1 += __shfl_xor(s1, off, 32);
  }
  if (lane == 0) { es[n * NHEAD + head] = s0; ed[n * NHEAD + head] = s1; }
}

// ---------------- CSR build ----------------
__global__ __launch_bounds__(256) void k_deg(const int* __restrict__ ei, int* __restrict__ deg) {
  int e = blockIdx.x * 256 + threadIdx.x;
  if (e >= ETOT) return;
  int dst = (e < NEDGES) ? ei[NEDGES + e] : (e - NEDGES);
  atomicAdd(&deg[dst], 1);
}

__global__ __launch_bounds__(1024) void k_scan(const int* __restrict__ deg, int* __restrict__ rowstart) {
  __shared__ int sd[1024];
  __shared__ int carry;
  if (threadIdx.x == 0) carry = 0;
  __syncthreads();
  for (int base = 0; base < NNODES; base += 1024) {
    int i = base + threadIdx.x;
    int v = (i < NNODES) ? deg[i] : 0;
    sd[threadIdx.x] = v;
    __syncthreads();
    for (int off = 1; off < 1024; off <<= 1) {
      int t = (threadIdx.x >= off) ? sd[threadIdx.x - off] : 0;
      __syncthreads();
      sd[threadIdx.x] += t;
      __syncthreads();
    }
    if (i < NNODES) rowstart[i] = carry + sd[threadIdx.x] - v;  // exclusive
    int tot = sd[1023];
    __syncthreads();
    if (threadIdx.x == 0) carry += tot;
    __syncthreads();
  }
  if (threadIdx.x == 0) rowstart[NNODES] = carry;
}

__global__ __launch_bounds__(256) void k_fill(const int* __restrict__ ei,
                                              const int* __restrict__ rowstart,
                                              int* __restrict__ cursor, int* __restrict__ csr) {
  int e = blockIdx.x * 256 + threadIdx.x;
  if (e >= ETOT) return;
  int src = (e < NEDGES) ? ei[e] : (e - NEDGES);
  int dst = (e < NEDGES) ? ei[NEDGES + e] : (e - NEDGES);
  int pos = atomicAdd(&cursor[dst], 1);
  csr[rowstart[dst] + pos] = src;
}

// ---------------- softmax + aggregate + relu(+bias) + pool-scatter ----------------
// one 256-thread block per destination node
__global__ __launch_bounds__(256) void k_agg(const float* __restrict__ H,
                                             const float* __restrict__ es,
                                             const float* __restrict__ ed,
                                             const int* __restrict__ rowstart,
                                             const int* __restrict__ csr,
                                             const float* __restrict__ bias,
                                             const int* __restrict__ batch,
                                             float* __restrict__ pool, int* __restrict__ cnt) {
  const int n = blockIdx.x, tid = threadIdx.x;
  const int start = rowstart[n], end = rowstart[n + 1];

  __shared__ float red[256];
  __shared__ float mh[NHEAD], sh[NHEAD], edn[NHEAD];
  if (tid < NHEAD) edn[tid] = ed[n * NHEAD + tid];
  __syncthreads();

  for (int hh = 0; hh < NHEAD; ++hh) {
    const float edv = edn[hh];
    float lm = -1e30f;
    for (int i = start + tid; i < end; i += 256) {
      float ev = es[csr[i] * NHEAD + hh] + edv;
      ev = ev > 0.f ? ev : NEG_SLOPE * ev;
      lm = fmaxf(lm, ev);
    }
    red[tid] = lm; __syncthreads();
    for (int off = 128; off; off >>= 1) { if (tid < off) red[tid] = fmaxf(red[tid], red[tid + off]); __syncthreads(); }
    float m = red[0];
    __syncthreads();
    float ls = 0.f;
    for (int i = start + tid; i < end; i += 256) {
      float ev = es[csr[i] * NHEAD + hh] + edv;
      ev = ev > 0.f ? ev : NEG_SLOPE * ev;
      ls += __expf(ev - m);
    }
    red[tid] = ls; __syncthreads();
    for (int off = 128; off; off >>= 1) { if (tid < off) red[tid] += red[tid + off]; __syncthreads(); }
    if (tid == 0) { mh[hh] = m; sh[hh] = red[0]; }
    __syncthreads();
  }

  const int CH = 64;
  __shared__ int   ssrc[CH];
  __shared__ float salpha[CH][NHEAD];
  float acc0 = 0.f, acc1 = 0.f, acc2 = 0.f, acc3 = 0.f;

  for (int base = start; base < end; base += CH) {
    int c = min(CH, end - base);
    if (tid < c) ssrc[tid] = csr[base + tid];
    __syncthreads();
    int i = tid >> 2, hh = tid & 3;    // 256 threads cover CH=64 edges x 4 heads
    if (i < c) {
      float ev = es[ssrc[i] * NHEAD + hh] + edn[hh];
      ev = ev > 0.f ? ev : NEG_SLOPE * ev;
      salpha[i][hh] = __expf(ev - mh[hh]) / (sh[hh] + 1e-16f);
    }
    __syncthreads();
    for (int j = 0; j < c; ++j) {
      const float* hrow = H + (size_t)ssrc[j] * DFEAT;
      float a0 = salpha[j][0], a1 = salpha[j][1], a2 = salpha[j][2], a3 = salpha[j][3];
      acc0 += a0 * hrow[tid];          // feature f = tid        -> head 0
      acc1 += a1 * hrow[tid + 256];    // head 1
      acc2 += a2 * hrow[tid + 512];    // head 2
      acc3 += a3 * hrow[tid + 768];    // head 3
    }
    __syncthreads();
  }

  const int g = batch[n];
  float* prow = pool + (size_t)g * DFEAT;
  float v0 = acc0 + bias[tid];        v0 = v0 > 0.f ? v0 : 0.f;
  float v1 = acc1 + bias[tid + 256];  v1 = v1 > 0.f ? v1 : 0.f;
  float v2 = acc2 + bias[tid + 512];  v2 = v2 > 0.f ? v2 : 0.f;
  float v3 = acc3 + bias[tid + 768];  v3 = v3 > 0.f ? v3 : 0.f;
  atomicAdd(&prow[tid],       v0);
  atomicAdd(&prow[tid + 256], v1);
  atomicAdd(&prow[tid + 512], v2);
  atomicAdd(&prow[tid + 768], v3);
  if (tid == 0) atomicAdd(&cnt[g], 1);
}

// ---------------- final: mean pool + [1024x2] projection ----------------
__global__ __launch_bounds__(256) void k_final(const float* __restrict__ pool,
                                               const int* __restrict__ cnt,
                                               const float* __restrict__ Wr,
                                               const float* __restrict__ br,
                                               float* __restrict__ out) {
  int g = blockIdx.x, tid = threadIdx.x;
  float c = fmaxf((float)cnt[g], 1.0f);
  float p0 = 0.f, p1 = 0.f;
  for (int d = tid; d < DFEAT; d += 256) {
    float mean = pool[(size_t)g * DFEAT + d] / c;
    p0 += mean * Wr[d * 2 + 0];
    p1 += mean * Wr[d * 2 + 1];
  }
  __shared__ float r0[256], r1[256];
  r0[tid] = p0; r1[tid] = p1; __syncthreads();
  for (int off = 128; off; off >>= 1) {
    if (tid < off) { r0[tid] += r0[tid + off]; r1[tid] += r1[tid + off]; }
    __syncthreads();
  }
  if (tid == 0) { out[g * 2 + 0] = r0[0] + br[0]; out[g * 2 + 1] = r1[0] + br[1]; }
}

// ---------------- launcher ----------------
extern "C" void kernel_launch(void* const* d_in, const int* in_sizes, int n_in,
                              void* d_out, int out_size, void* d_ws, size_t ws_size,
                              hipStream_t stream) {
  const float* x     = (const float*)d_in[0];
  const int*   ei    = (const int*)  d_in[1];
  const int*   batch = (const int*)  d_in[2];
  const float* W     = (const float*)d_in[3];   // (NB, 1024, 1024)
  const float* a_src = (const float*)d_in[4];   // (NB, 4, 256)
  const float* a_dst = (const float*)d_in[5];
  const float* b     = (const float*)d_in[6];   // (NB, 1024)
  const float* Wr    = (const float*)d_in[7];   // (1024, 2)
  const float* br    = (const float*)d_in[8];   // (2,)
  float* out = (float*)d_out;

  // only block 2 is live (each block reads the ORIGINAL x; earlier outputs overwritten)
  const float* W2  = W     + (size_t)2 * DFEAT * DFEAT;
  const float* as2 = a_src + (size_t)2 * NHEAD * DHEAD;
  const float* ad2 = a_dst + (size_t)2 * NHEAD * DHEAD;
  const float* b2  = b     + (size_t)2 * DFEAT;

  // workspace carve-up (~65 MB)
  char* ws = (char*)d_ws;
  size_t off = 0;
  auto carve = [&](size_t bytes) -> void* {
    void* p = ws + off;
    off = (off + bytes + 255) & ~(size_t)255;
    return p;
  };
  __bf16* xb   = (__bf16*)carve((size_t)NNODES * DFEAT * 2 + 8192);  // +pad for pipeline over-read
  __bf16* wt   = (__bf16*)carve((size_t)DFEAT * DFEAT * 2 + 8192);   // +pad for pipeline over-read
  float*  H    = (float*) carve((size_t)NNODES * DFEAT * 4);
  float*  es   = (float*) carve((size_t)NNODES * NHEAD * 4);
  float*  ed   = (float*) carve((size_t)NNODES * NHEAD * 4);
  int*    deg  = (int*)   carve((size_t)NNODES * 4);
  int*    rows = (int*)   carve((size_t)(NNODES + 1) * 4);
  int*    cur  = (int*)   carve((size_t)NNODES * 4);
  int*    csr  = (int*)   carve((size_t)ETOT * 4);
  float*  pool = (float*) carve((size_t)GGRAPH * DFEAT * 4);
  int*    cnt  = (int*)   carve((size_t)GGRAPH * 4);

  hipMemsetAsync(deg,  0, (size_t)NNODES * 4, stream);
  hipMemsetAsync(cur,  0, (size_t)NNODES * 4, stream);
  hipMemsetAsync(pool, 0, (size_t)GGRAPH * DFEAT * 4, stream);
  hipMemsetAsync(cnt,  0, (size_t)GGRAPH * 4, stream);

  k_cvt_x<<<(NNODES * DFEAT + 255) / 256, 256, 0, stream>>>(x, xb);
  k_cvt_w<<<(DFEAT * DFEAT + 255) / 256, 256, 0, stream>>>(W2, wt);
  k_gemm <<<(NNODES / 16) * 2, 256, 0, stream>>>(xb, wt, H);
  k_logits<<<NNODES, 128, 0, stream>>>(H, as2, ad2, es, ed);
  k_deg  <<<(ETOT + 255) / 256, 256, 0, stream>>>(ei, deg);
  k_scan <<<1, 1024, 0, stream>>>(deg, rows);
  k_fill <<<(ETOT + 255) / 256, 256, 0, stream>>>(ei, rows, cur, csr);
  k_agg  <<<NNODES, 256, 0, stream>>>(H, es, ed, rows, csr, b2, batch, pool, cnt);
  k_final<<<GGRAPH, 256, 0, stream>>>(pool, cnt, Wr, br, out);
}
